// Model_86655260164949
// MI455X (gfx1250) — matrix operand order repaired and checked by reading, not compiled, not run on hardware
//
#include <hip/hip_runtime.h>
#include <hip/hip_bf16.h>

// ---------------- model constants ----------------
#define DMODEL 512
#define NHEAD  8
#define HDIM   64
#define FFDIM  2048
#define BATCH  512
#define SO_LEN 32
#define SP_LEN 32
#define PPH    8
#define ORTHV  128
#define PHONV  64
#define TMIX   65      // 1 + SO + SP

#define USE_ASYNC 1    // flip to 0 if the async-to-LDS mnemonic is rejected

typedef __attribute__((ext_vector_type(16))) __bf16        v16bf;
typedef __attribute__((ext_vector_type(8)))  float         v8f;
typedef __attribute__((ext_vector_type(8)))  unsigned int  v8u;

enum { GEMM_RELU = 1, GEMM_BIAS = 2, GEMM_RES = 4 };

// pack two f32 -> two bf16 in one dword (lo in [15:0], hi in [31:16]), RNE
static __device__ __forceinline__ unsigned pack2bf(float lo, float hi) {
  union { float f; unsigned u; } a, b; a.f = lo; b.f = hi;
  unsigned rl = ((a.u + 0x7FFFu + ((a.u >> 16) & 1u)) >> 16) & 0xFFFFu;
  unsigned rh =  (b.u + 0x7FFFu + ((b.u >> 16) & 1u)) & 0xFFFF0000u;
  return rl | rh;
}

static __device__ __forceinline__ v16bf as_v16bf(v8u u) {
  v16bf r; __builtin_memcpy(&r, &u, 32); return r;
}

static __device__ __forceinline__ void wait_async0() {
#if __has_builtin(__builtin_amdgcn_s_wait_asynccnt)
  __builtin_amdgcn_s_wait_asynccnt(0);
#else
  asm volatile("s_wait_asynccnt 0x0" ::: "memory");
#endif
}

#if USE_ASYNC
#define ASYNC_B128(ldsaddr, gaddr, off_) \
  asm volatile("global_load_async_to_lds_b128 %0, %1, off offset:" #off_ \
               :: "v"(ldsaddr), "v"(gaddr) : "memory")
#endif

// ---------------- operand pre-conversion ----------------
// A: f32 [M,K] (row stride lda) -> dense bf16 [M,K]
__global__ void cvt_a_kernel(const float* __restrict__ src, long lda,
                             unsigned* __restrict__ dst, long Kd, long total8)
{
  long i = (long)blockIdx.x * blockDim.x + threadIdx.x;
  if (i >= total8) return;
  long e = i * 8;
  long row = e / Kd, col = e % Kd;
  const float4* s = (const float4*)(src + row * lda + col);
  float4 x0 = s[0], x1 = s[1];
  uint4 w;
  w.x = pack2bf(x0.x, x0.y); w.y = pack2bf(x0.z, x0.w);
  w.z = pack2bf(x1.x, x1.y); w.w = pack2bf(x1.z, x1.w);
  ((uint4*)dst)[i] = w;
}

// B: f32 [K,N] dense -> transposed pair-major dwords Bt[N][K/2]
// Bt[n][p] = { bf16(B[2p][n]) , bf16(B[2p+1][n]) }  (lo, hi)
__global__ void cvt_b_tr_kernel(const float* __restrict__ src, int N, int K,
                                unsigned* __restrict__ dst, long total)
{
  long i = (long)blockIdx.x * blockDim.x + threadIdx.x;
  if (i >= total) return;                 // total = N * (K/2); i = n*(K/2) + p
  long kh = (long)(K >> 1);
  int  p  = (int)(i % kh);
  long n  = i / kh;
  float lo = src[(long)(2 * p)     * N + n];
  float hi = src[(long)(2 * p + 1) * N + n];
  dst[i] = pack2bf(lo, hi);
}

// ---------------- WMMA GEMM on pre-converted operands ----------------
// Ab: bf16 dense [M,K]; Bt: pair-major dwords [N][K/2] (row stride Kh=K/2).
// Requirements: M%64==0, N%128==0, K%32==0.
// block = 128 threads (4 waves), tile 64(M) x 128(N), K-step 32.
// Wave (wm,wn) owns a 32x64 patch -> 2x4 v_wmma_f32_16x16x32_bf16 per K-step.
__global__ __launch_bounds__(128)
void gemm_bf16_kernel(const __bf16* __restrict__ Ab, long Ka,
                      const unsigned* __restrict__ Bt, long Kh,
                      const float* __restrict__ bias,
                      const float* __restrict__ res, long resStride,
                      float* __restrict__ C, long ldc,
                      int K, int flags)
{
  __shared__ __align__(16) __bf16    As[64][40];    // 80B rows
  __shared__ __align__(16) unsigned  BsT[128][20];  // one N-column per row, 80B rows

  const int tid  = threadIdx.x;
  const int lane = tid & 31;
  const int wave = tid >> 5;
  const int wm   = wave >> 1;        // 0..1 : 32-row half
  const int wn   = wave & 1;         // 0..1 : 64-col half
  const int m0   = blockIdx.y * 64;
  const int n0   = blockIdx.x * 128;
  const int lrow = lane & 15;
  const int hi   = lane >> 4;

  const int arow = tid >> 1, acol = (tid & 1) * 16;   // A: 16 bf16 (32B) / thread

  const __bf16*   ga = Ab + (long)(m0 + arow) * Ka + acol;   // +32 elems per K-step
  const unsigned* gb = Bt + (long)(n0 + tid) * Kh;           // +16 dwords per K-step

  const unsigned ldsA = (unsigned)(unsigned long long)&As[arow][acol];
  const unsigned ldsB = (unsigned)(unsigned long long)&BsT[tid][0];

  v8f acc[2][4];
  #pragma unroll
  for (int i = 0; i < 2; ++i)
    #pragma unroll
    for (int j = 0; j < 4; ++j)
      #pragma unroll
      for (int r = 0; r < 8; ++r) acc[i][j][r] = 0.0f;

  #pragma unroll 2
  for (int k0 = 0; k0 < K; k0 += 32) {
#if USE_ASYNC
    ASYNC_B128(ldsA, ga, 0);
    ASYNC_B128(ldsA, ga, 16);
    ASYNC_B128(ldsB, gb, 0);
    ASYNC_B128(ldsB, gb, 16);
    ASYNC_B128(ldsB, gb, 32);
    ASYNC_B128(ldsB, gb, 48);
    wait_async0();
#else
    {
      const uint4* gaa = (const uint4*)ga;
      uint4 t0 = gaa[0], t1 = gaa[1];
      *(uint4*)&As[arow][acol]     = t0;
      *(uint4*)&As[arow][acol + 8] = t1;
      const uint4* gbb = (const uint4*)gb;
      #pragma unroll
      for (int s = 0; s < 4; ++s) *(uint4*)&BsT[tid][4 * s] = gbb[s];
    }
#endif
    ga += 32;
    gb += 16;
    __syncthreads();

    // ---- fragments per CDNA5 16-bit WMMA VGPR layouts ----
    v16bf af[2], bfr[4];
    #pragma unroll
    for (int i = 0; i < 2; ++i) {
      const int ar = wm * 32 + i * 16 + lrow;       // A row (M)
      const int kb = hi * 8;                        // K base per lane group
      uint4 lo4 = *(const uint4*)&As[ar][kb];       // K = kb .. kb+7
      uint4 hi4 = *(const uint4*)&As[ar][kb + 16];  // K = kb+16 .. kb+23
      v8u au;
      au[0] = lo4.x; au[1] = lo4.y; au[2] = lo4.z; au[3] = lo4.w;
      au[4] = hi4.x; au[5] = hi4.y; au[6] = hi4.z; au[7] = hi4.w;
      af[i] = as_v16bf(au);
    }
    #pragma unroll
    for (int j = 0; j < 4; ++j) {
      const int bc  = wn * 64 + j * 16 + lrow;      // B col (N)
      const int kbp = hi * 4;                       // K-pair base per lane group
      uint4 lo4 = *(const uint4*)&BsT[bc][kbp];     // pairs kbp .. kbp+3
      uint4 hi4 = *(const uint4*)&BsT[bc][kbp + 8]; // pairs kbp+8 .. kbp+11
      v8u bu;
      bu[0] = lo4.x; bu[1] = lo4.y; bu[2] = lo4.z; bu[3] = lo4.w;
      bu[4] = hi4.x; bu[5] = hi4.y; bu[6] = hi4.z; bu[7] = hi4.w;
      bfr[j] = as_v16bf(bu);
    }
    #pragma unroll
    for (int i = 0; i < 2; ++i)
      #pragma unroll
      for (int j = 0; j < 4; ++j)
        acc[i][j] = __builtin_amdgcn_wmma_f32_16x16x32_bf16(
            false, af[i], false, bfr[j], (short)0, acc[i][j], false, false);
    __syncthreads();
  }

  // epilogue: bias / relu / residual fused; C/D layout: VGPR r -> M = r + 8*hi
  #pragma unroll
  for (int i = 0; i < 2; ++i) {
    #pragma unroll
    for (int j = 0; j < 4; ++j) {
      const int gn = n0 + wn * 64 + j * 16 + lrow;
      const int gm0 = m0 + wm * 32 + i * 16 + hi * 8;
      #pragma unroll
      for (int r = 0; r < 8; ++r) {
        const int gm = gm0 + r;
        float vv = acc[i][j][r];
        if (flags & GEMM_BIAS) vv += bias[gn];
        if (flags & GEMM_RELU) vv = fmaxf(vv, 0.0f);
        if (flags & GEMM_RES)  vv += res[(long)gm * resStride + gn];
        C[(long)gm * ldc + gn] = vv;
      }
    }
  }
}

// ---------------- attention (scores+softmax+PV), one block per (b,h,q) ----------------
__global__ __launch_bounds__(64)
void attn_kernel(const float* __restrict__ q, const float* __restrict__ k,
                 const float* __restrict__ v, float* __restrict__ out,
                 int Tq, int Tk, const unsigned char* kpm, int causal)
{
  const int b = blockIdx.z, h = blockIdx.y, qi = blockIdx.x;
  const int tid = threadIdx.x;
  __shared__ float sc[TMIX + 7];
  const float* qr = q + ((long)b * Tq + qi) * DMODEL + h * HDIM;
  for (int kk = tid; kk < Tk; kk += 64) {
    const float* kr = k + ((long)b * Tk + kk) * DMODEL + h * HDIM;
    float s = 0.0f;
    #pragma unroll 8
    for (int d = 0; d < HDIM; ++d) s += qr[d] * kr[d];
    s *= 0.125f;                                  // 1/sqrt(64)
    if (causal && kk > qi) s = -1e9f;
    if (kpm && kpm[(long)b * Tk + kk]) s = -1e9f;
    sc[kk] = s;
  }
  __syncthreads();
  float mx = -3.0e38f;
  for (int kk = 0; kk < Tk; ++kk) mx = fmaxf(mx, sc[kk]);
  __syncthreads();
  for (int kk = tid; kk < Tk; kk += 64) sc[kk] = __expf(sc[kk] - mx);
  __syncthreads();
  float sum = 0.0f;
  for (int kk = 0; kk < Tk; ++kk) sum += sc[kk];
  const float inv = 1.0f / sum;
  float a = 0.0f;
  for (int kk = 0; kk < Tk; ++kk)
    a += sc[kk] * v[((long)b * Tk + kk) * DMODEL + h * HDIM + tid];
  out[((long)b * Tq + qi) * DMODEL + h * HDIM + tid] = a * inv;
}

// ---------------- layernorm over D=512, one block per row ----------------
__global__ __launch_bounds__(256)
void layernorm_kernel(const float* __restrict__ x, const float* __restrict__ g,
                      const float* __restrict__ bta, float* __restrict__ y)
{
  const int row = blockIdx.x;
  const int tid = threadIdx.x;
  __shared__ float red[256];
  const float* xr = x + (long)row * DMODEL;
  float a0 = xr[tid], a1 = xr[tid + 256];
  red[tid] = a0 + a1;
  __syncthreads();
  for (int o = 128; o > 0; o >>= 1) { if (tid < o) red[tid] += red[tid + o]; __syncthreads(); }
  const float mean = red[0] * (1.0f / DMODEL);
  __syncthreads();
  const float d0 = a0 - mean, d1 = a1 - mean;
  red[tid] = d0 * d0 + d1 * d1;
  __syncthreads();
  for (int o = 128; o > 0; o >>= 1) { if (tid < o) red[tid] += red[tid + o]; __syncthreads(); }
  const float rstd = rsqrtf(red[0] * (1.0f / DMODEL) + 1e-5f);
  float* yr = y + (long)row * DMODEL;
  yr[tid]       = d0 * rstd * g[tid]       + bta[tid];
  yr[tid + 256] = d1 * rstd * g[tid + 256] + bta[tid + 256];
}

// ---------------- embeddings ----------------
__global__ void embed_orth_kernel(const int* __restrict__ tok, const float* __restrict__ emb,
                                  const float* __restrict__ pos, float* __restrict__ out, int S)
{
  long idx = (long)blockIdx.x * blockDim.x + threadIdx.x;
  long total = (long)BATCH * S * DMODEL;
  if (idx >= total) return;
  int d = (int)(idx % DMODEL);
  long bs = idx / DMODEL;
  int s = (int)(bs % S);
  int t = tok[bs];
  out[idx] = emb[(long)t * DMODEL + d] + pos[(long)s * DMODEL + d];
}

__global__ void embed_phon_kernel(const int* __restrict__ tok, const float* __restrict__ emb,
                                  const float* __restrict__ pos, float* __restrict__ out, int S)
{
  long idx = (long)blockIdx.x * blockDim.x + threadIdx.x;
  long total = (long)BATCH * S * DMODEL;
  if (idx >= total) return;
  int d = (int)(idx % DMODEL);
  long bs = idx / DMODEL;
  int s = (int)(bs % S);
  const int* tp = tok + bs * PPH;
  float acc = 0.0f;
  #pragma unroll
  for (int j = 0; j < PPH; ++j) acc += emb[(long)tp[j] * DMODEL + d];
  out[idx] = acc * (1.0f / PPH) + pos[(long)s * DMODEL + d];
}

// ---------------- mix-sequence concat: [global | gp+o | pg+ph] ----------------
__global__ void build_mix_kernel(const float* __restrict__ g, const float* __restrict__ gp,
                                 const float* __restrict__ o, const float* __restrict__ pg,
                                 const float* __restrict__ ph, float* __restrict__ out)
{
  long idx = (long)blockIdx.x * blockDim.x + threadIdx.x;
  long total = (long)BATCH * TMIX * DMODEL;
  if (idx >= total) return;
  int d = (int)(idx % DMODEL);
  long bt = idx / DMODEL;
  int t = (int)(bt % TMIX);
  long b = bt / TMIX;
  float val;
  if (t == 0) {
    val = g[d];
  } else if (t <= SO_LEN) {
    long off = ((long)b * SO_LEN + (t - 1)) * DMODEL + d;
    val = gp[off] + o[off];
  } else {
    long off = ((long)b * SP_LEN + (t - 1 - SO_LEN)) * DMODEL + d;
    val = pg[off] + ph[off];
  }
  out[idx] = val;
}

__global__ void build_mask_kernel(const unsigned char* __restrict__ oem,
                                  const unsigned char* __restrict__ pem,
                                  unsigned char* __restrict__ out)
{
  int idx = blockIdx.x * blockDim.x + threadIdx.x;
  if (idx >= BATCH * TMIX) return;
  int t = idx % TMIX;
  int b = idx / TMIX;
  unsigned char m = 0;
  if (t >= 1 && t <= SO_LEN) m = oem[b * SO_LEN + (t - 1)];
  else if (t > SO_LEN)       m = pem[b * SP_LEN + (t - 1 - SO_LEN)];
  out[idx] = m;
}

// ---------------- pad phonology head to N=128 ----------------
__global__ void pad_head_kernel(const float* __restrict__ W, const float* __restrict__ b,
                                float* __restrict__ Wp, float* __restrict__ bp)
{
  int idx = blockIdx.x * blockDim.x + threadIdx.x;   // over DMODEL*128
  if (idx < DMODEL * 128) {
    int n = idx & 127, r = idx >> 7;
    Wp[idx] = (n < 2 * (PHONV - 1)) ? W[r * (2 * (PHONV - 1)) + n] : 0.0f;
  }
  if (idx < 128) bp[idx] = (idx < 2 * (PHONV - 1)) ? b[idx] : 0.0f;
}

// ---------------- output head reshapes ----------------
__global__ void transpose_ol_kernel(const float* __restrict__ in, float* __restrict__ out)
{
  long idx = (long)blockIdx.x * blockDim.x + threadIdx.x;
  long total = (long)BATCH * ORTHV * SO_LEN;
  if (idx >= total) return;
  int s = (int)(idx % SO_LEN);
  long t = idx / SO_LEN;
  int vv = (int)(t % ORTHV);
  long b = t / ORTHV;
  out[idx] = in[((long)b * SO_LEN + s) * ORTHV + vv];
}

// input has padded row stride 128; logical row = [2*(PHONV-1)] = 126
__global__ void reshape_pl_kernel(const float* __restrict__ in, float* __restrict__ out)
{
  long idx = (long)blockIdx.x * blockDim.x + threadIdx.x;
  const int VH = PHONV - 1;                       // 63
  long total = (long)BATCH * 2 * SP_LEN * VH;
  if (idx >= total) return;
  int j = (int)(idx % VH);
  long t = idx / VH;
  int s = (int)(t % SP_LEN);
  long t2 = t / SP_LEN;
  int c = (int)(t2 % 2);
  long b = t2 / 2;
  out[idx] = in[((long)b * SP_LEN + s) * 128 + c * VH + j];
}

// ================= host-side orchestration =================
struct MhaP { const float *Wq, *Wk, *Wv, *Wo, *bq, *bk, *bv, *bo; };
struct Scratch { float *q, *k, *v, *ctx, *tmp, *hid; };

static unsigned* g_abf = nullptr;   // A pre-converted (bf16 as dwords)
static unsigned* g_bpb = nullptr;   // B pre-converted (transposed pair-major dwords)

// Requires M%64==0, N%128==0, K%32==0, lda%4==0 (true for every call site).
static void run_gemm(hipStream_t st, const float* A, long lda, const float* Bm, long /*ldb==N*/,
                     const float* bias, const float* res, long resStride,
                     float* C, long ldc, int M, int N, int K, int flags)
{
  const long tot8 = (long)M * K / 8;
  cvt_a_kernel<<<(unsigned)((tot8 + 255) / 256), 256, 0, st>>>(A, lda, g_abf, K, tot8);
  const long totB = (long)N * (K / 2);
  cvt_b_tr_kernel<<<(unsigned)((totB + 255) / 256), 256, 0, st>>>(Bm, N, K, g_bpb, totB);
  dim3 g(N / 128, M / 64);
  gemm_bf16_kernel<<<g, 128, 0, st>>>((const __bf16*)g_abf, K, g_bpb, K / 2,
                                      bias, res, resStride, C, ldc, K, flags);
}

static void run_ln(hipStream_t st, const float* x, const float* g_, const float* b_, float* y, int rows)
{
  layernorm_kernel<<<rows, 256, 0, st>>>(x, g_, b_, y);
}

static void run_mha(hipStream_t st, const float* xq, int Tq, const float* xkv, int Tk,
                    const MhaP& p, const unsigned char* kpm, int causal,
                    const float* residual, float* out, const Scratch& S)
{
  const int Mq = BATCH * Tq, Mk = BATCH * Tk;
  run_gemm(st, xq,  DMODEL, p.Wq, DMODEL, p.bq, nullptr, 0, S.q, DMODEL, Mq, DMODEL, DMODEL, GEMM_BIAS);
  run_gemm(st, xkv, DMODEL, p.Wk, DMODEL, p.bk, nullptr, 0, S.k, DMODEL, Mk, DMODEL, DMODEL, GEMM_BIAS);
  run_gemm(st, xkv, DMODEL, p.Wv, DMODEL, p.bv, nullptr, 0, S.v, DMODEL, Mk, DMODEL, DMODEL, GEMM_BIAS);
  dim3 g(Tq, NHEAD, BATCH);
  attn_kernel<<<g, 64, 0, st>>>(S.q, S.k, S.v, S.ctx, Tq, Tk, kpm, causal);
  run_gemm(st, S.ctx, DMODEL, p.Wo, DMODEL, p.bo, residual, DMODEL, out, DMODEL, Mq, DMODEL, DMODEL,
           GEMM_BIAS | (residual ? GEMM_RES : 0));
}

// enc layer leaves: [0..7]=attn Wq,Wk,Wv,Wo,bq,bk,bv,bo  [8,9]=ln1  [10..13]=W1,b1,W2,b2  [14,15]=ln2
static void run_enc_layer(hipStream_t st, float* x, int T, const float* const* L,
                          const unsigned char* kpm, const Scratch& S)
{
  MhaP a{L[0], L[1], L[2], L[3], L[4], L[5], L[6], L[7]};
  const int M = BATCH * T;
  run_mha(st, x, T, x, T, a, kpm, 0, x, S.tmp, S);
  run_ln(st, S.tmp, L[8], L[9], x, M);
  run_gemm(st, x, DMODEL, L[10], FFDIM, L[11], nullptr, 0, S.hid, FFDIM, M, FFDIM, DMODEL, GEMM_BIAS | GEMM_RELU);
  run_gemm(st, S.hid, FFDIM, L[12], DMODEL, L[13], x, DMODEL, S.tmp, DMODEL, M, DMODEL, FFDIM, GEMM_BIAS | GEMM_RES);
  run_ln(st, S.tmp, L[14], L[15], x, M);
}

// dec leaves: [0..7]=sa [8,9]=ln1 [10..17]=ca [18,19]=ln2 [20..23]=W1,b1,W2,b2 [24,25]=ln3
static void run_dec_layer(hipStream_t st, float* x, int T, const float* mem,
                          const float* const* L, const unsigned char* kpm, const Scratch& S)
{
  const int M = BATCH * T;
  MhaP sa{L[0], L[1], L[2], L[3], L[4], L[5], L[6], L[7]};
  run_mha(st, x, T, x, T, sa, kpm, 1, x, S.tmp, S);
  run_ln(st, S.tmp, L[8], L[9], x, M);
  MhaP ca{L[10], L[11], L[12], L[13], L[14], L[15], L[16], L[17]};
  run_mha(st, x, T, mem, 1, ca, nullptr, 0, x, S.tmp, S);
  run_ln(st, S.tmp, L[18], L[19], x, M);
  run_gemm(st, x, DMODEL, L[20], FFDIM, L[21], nullptr, 0, S.hid, FFDIM, M, FFDIM, DMODEL, GEMM_BIAS | GEMM_RELU);
  run_gemm(st, S.hid, FFDIM, L[22], DMODEL, L[23], x, DMODEL, S.tmp, DMODEL, M, DMODEL, FFDIM, GEMM_BIAS | GEMM_RES);
  run_ln(st, S.tmp, L[24], L[25], x, M);
}

extern "C" void kernel_launch(void* const* d_in, const int* in_sizes, int n_in,
                              void* d_out, int out_size, void* d_ws, size_t ws_size,
                              hipStream_t stream)
{
  (void)in_sizes; (void)n_in; (void)out_size; (void)ws_size;

  // ---- param leaves in setup_inputs() insertion order ----
  int pi = 0;
  auto F = [&](int i) { return (const float*)d_in[i]; };
  const float* orth_emb = F(pi++);
  const float* orth_pos = F(pi++);
  const float* phon_emb = F(pi++);
  const float* phon_pos = F(pi++);
  const float* gvec     = F(pi++);
  const float* oencL[16]; for (int i = 0; i < 16; ++i) oencL[i] = F(pi++);
  const float* pencL[16]; for (int i = 0; i < 16; ++i) pencL[i] = F(pi++);
  const float* mixL[16];  for (int i = 0; i < 16; ++i) mixL[i]  = F(pi++);
  const float* gpL[8];    for (int i = 0; i < 8;  ++i) gpL[i]   = F(pi++);
  const float* gp_ln[2];  for (int i = 0; i < 2;  ++i) gp_ln[i] = F(pi++);
  const float* pgL[8];    for (int i = 0; i < 8;  ++i) pgL[i]   = F(pi++);
  const float* pg_ln[2];  for (int i = 0; i < 2;  ++i) pg_ln[i] = F(pi++);
  const float* red_W = F(pi++);
  const float* red_b = F(pi++);
  const float* red_ln[2]; for (int i = 0; i < 2;  ++i) red_ln[i] = F(pi++);
  const float* odecL[26]; for (int i = 0; i < 26; ++i) odecL[i] = F(pi++);
  const float* pdecL[26]; for (int i = 0; i < 26; ++i) pdecL[i] = F(pi++);
  const float* oh_W = F(pi++);
  const float* oh_b = F(pi++);
  const float* ph_W = F(pi++);
  const float* ph_b = F(pi++);

  const int* oe_tok = (const int*)d_in[pi++];
  const unsigned char* oem = (const unsigned char*)d_in[pi++];
  const int* od_tok = (const int*)d_in[pi++];
  const unsigned char* odm = (const unsigned char*)d_in[pi++];
  const int* pe_tok = (const int*)d_in[pi++];
  const unsigned char* pem = (const unsigned char*)d_in[pi++];
  const int* pd_tok = (const int*)d_in[pi++];
  const unsigned char* pdm = (const unsigned char*)d_in[pi++];

  // ---- workspace bump allocator ----
  char* base = (char*)d_ws;
  size_t off = 0;
  auto allocB = [&](size_t bytes) -> void* {
    void* p = base + off;
    off += (bytes + 255) & ~(size_t)255;
    return p;
  };
  float* o    = (float*)allocB((size_t)BATCH * SO_LEN * DMODEL * 4);
  float* ph   = (float*)allocB((size_t)BATCH * SP_LEN * DMODEL * 4);
  Scratch S;
  S.q   = (float*)allocB((size_t)BATCH * TMIX * DMODEL * 4);
  S.k   = (float*)allocB((size_t)BATCH * TMIX * DMODEL * 4);
  S.v   = (float*)allocB((size_t)BATCH * TMIX * DMODEL * 4);
  S.ctx = (float*)allocB((size_t)BATCH * TMIX * DMODEL * 4);
  S.tmp = (float*)allocB((size_t)BATCH * TMIX * DMODEL * 4);
  S.hid = (float*)allocB((size_t)BATCH * TMIX * FFDIM * 4);
  float* mixx = (float*)allocB((size_t)BATCH * TMIX * DMODEL * 4);
  float* fin  = (float*)allocB((size_t)BATCH * DMODEL * 4);
  float* phWp = (float*)allocB((size_t)DMODEL * 128 * 4);   // padded phon head weight
  float* phbp = (float*)allocB((size_t)128 * 4);            // padded phon head bias
  g_abf = (unsigned*)allocB((size_t)BATCH * TMIX * FFDIM * 2);      // max M*K bf16
  g_bpb = (unsigned*)allocB((size_t)(FFDIM / 2) * DMODEL * 4);      // max N*(K/2) dwords
  unsigned char* m65 = (unsigned char*)allocB((size_t)BATCH * TMIX);

  const long totO = (long)BATCH * SO_LEN * DMODEL;
  const long totP = (long)BATCH * SP_LEN * DMODEL;
  const long totM = (long)BATCH * TMIX * DMODEL;

  // ---- pad phonology head (makes every GEMM in the model fully tile-aligned) ----
  pad_head_kernel<<<(DMODEL * 128 + 255) / 256, 256, 0, stream>>>(ph_W, ph_b, phWp, phbp);

  // ---- encoders ----
  embed_orth_kernel<<<(totO + 255) / 256, 256, 0, stream>>>(oe_tok, orth_emb, orth_pos, o, SO_LEN);
  run_enc_layer(stream, o, SO_LEN, oencL, oem, S);
  embed_phon_kernel<<<(totP + 255) / 256, 256, 0, stream>>>(pe_tok, phon_emb, phon_pos, ph, SP_LEN);
  run_enc_layer(stream, ph, SP_LEN, pencL, pem, S);

  // ---- cross mixing: gp / pg ----
  MhaP gpP{gpL[0], gpL[1], gpL[2], gpL[3], gpL[4], gpL[5], gpL[6], gpL[7]};
  run_mha(stream, o, SO_LEN, ph, SP_LEN, gpP, pem, 0, o, S.tmp, S);     // gp -> S.tmp
  run_ln(stream, S.tmp, gp_ln[0], gp_ln[1], S.tmp, BATCH * SO_LEN);
  MhaP pgP{pgL[0], pgL[1], pgL[2], pgL[3], pgL[4], pgL[5], pgL[6], pgL[7]};
  run_mha(stream, ph, SP_LEN, o, SO_LEN, pgP, oem, 0, ph, S.hid, S);    // pg -> S.hid
  run_ln(stream, S.hid, pg_ln[0], pg_ln[1], S.hid, BATCH * SP_LEN);

  build_mix_kernel<<<(totM + 255) / 256, 256, 0, stream>>>(gvec, S.tmp, o, S.hid, ph, mixx);
  build_mask_kernel<<<(BATCH * TMIX + 255) / 256, 256, 0, stream>>>(oem, pem, m65);
  run_enc_layer(stream, mixx, TMIX, mixL, m65, S);

  // ---- reduction to memory token: fin = LN(mixed[:,0] @ red_W + red_b + global) ----
  run_gemm(stream, mixx, (long)TMIX * DMODEL, red_W, DMODEL, red_b, gvec, 0,
           fin, DMODEL, BATCH, DMODEL, DMODEL, GEMM_BIAS | GEMM_RES);
  run_ln(stream, fin, red_ln[0], red_ln[1], fin, BATCH);

  // ---- decoders (reuse o / ph buffers) ----
  embed_orth_kernel<<<(totO + 255) / 256, 256, 0, stream>>>(od_tok, orth_emb, orth_pos, o, SO_LEN);
  run_dec_layer(stream, o, SO_LEN, fin, odecL, odm, S);
  embed_phon_kernel<<<(totP + 255) / 256, 256, 0, stream>>>(pd_tok, phon_emb, phon_pos, ph, SP_LEN);
  run_dec_layer(stream, ph, SP_LEN, fin, pdecL, pdm, S);

  // ---- output heads ----
  float* lo = S.q;   // [B*SO, ORTHV]
  run_gemm(stream, o, DMODEL, oh_W, ORTHV, oh_b, nullptr, 0, lo, ORTHV,
           BATCH * SO_LEN, ORTHV, DMODEL, GEMM_BIAS);
  const long totOL = (long)BATCH * ORTHV * SO_LEN;
  transpose_ol_kernel<<<(totOL + 255) / 256, 256, 0, stream>>>(lo, (float*)d_out);

  float* lp = S.k;   // [B*SP, 128] (padded)
  run_gemm(stream, ph, DMODEL, phWp, 128, phbp, nullptr, 0, lp, 128,
           BATCH * SP_LEN, 128, DMODEL, GEMM_BIAS);
  const long totPL = (long)BATCH * 2 * SP_LEN * (PHONV - 1);
  reshape_pl_kernel<<<(totPL + 255) / 256, 256, 0, stream>>>(lp, (float*)d_out + totOL);
}
